// GCN_7894149890262
// MI455X (gfx1250) — matrix operand order repaired and checked by reading, not compiled, not run on hardware
//
#include <hip/hip_runtime.h>
#include <hip/hip_bf16.h>

#define GN 100000
#define GD 512
#define GE 3200000

typedef __attribute__((ext_vector_type(16))) __bf16          v16bf;
typedef __attribute__((ext_vector_type(8)))  float           v8f;
typedef __attribute__((ext_vector_type(8)))  unsigned short  v8us;
typedef __attribute__((ext_vector_type(16))) unsigned short  v16us;
typedef __attribute__((ext_vector_type(4)))  unsigned int    v4u;
typedef __attribute__((ext_vector_type(4)))  int             v4i;
typedef __attribute__((ext_vector_type(8)))  int             v8i;

#if defined(__has_builtin)
# if __has_builtin(__builtin_amdgcn_tensor_load_to_lds)
#  define HAVE_TDM 1
# else
#  define HAVE_TDM 0
# endif
#else
# define HAVE_TDM 0
#endif

// round-to-nearest-even f32 -> bf16 (bit trick)
static __device__ __forceinline__ unsigned short f2bf(float f) {
  union { float f; unsigned int u; } v; v.f = f;
  unsigned int u = v.u;
  u += 0x7FFFu + ((u >> 16) & 1u);
  return (unsigned short)(u >> 16);
}

static __device__ __forceinline__ unsigned long long pack4bf(float4 f) {
  return  (unsigned long long)f2bf(f.x)
       | ((unsigned long long)f2bf(f.y) << 16)
       | ((unsigned long long)f2bf(f.z) << 32)
       | ((unsigned long long)f2bf(f.w) << 48);
}

union BFU16 { v16us u; v16bf b; };

static __device__ __forceinline__ v16bf mk16(v8us lo, v8us hi) {
  BFU16 t;
  #pragma unroll
  for (int i = 0; i < 8; ++i) { t.u[i] = lo[i]; t.u[i + 8] = hi[i]; }
  return t.b;
}

#define LDA 40   // ushort stride: 64B data + 16B pad = 80B rows (TDM pad fields)
#define LDB 40

#if HAVE_TDM
// Issue one 2D TDM tile load: global (row-major, stride0 elements of 2B)
// -> LDS, HW-padded 16B per 64B row so LDS stride is 40 ushorts.
// D# packing per CDNA5 ISA.
static __device__ __forceinline__ void tdm_load_2d(
    unsigned lds_addr, const void* gp,
    unsigned tensor_d0, unsigned tensor_d1,
    unsigned tile_d0, unsigned tile_d1, unsigned stride0) {
  unsigned long long ga = (unsigned long long)(size_t)gp;
  v4u g0 = { 1u,                                   // count=1, user desc
             lds_addr,                             // [63:32]  LDS byte addr
             (unsigned)ga,                         // [95:64]  global addr lo
             (unsigned)((ga >> 32) & 0x1FFFFFFu)   // [120:96] global addr hi
             | 0x80000000u };                      // [127:126] type=2 (image)
  v8i g1 = { (int)((1u << 16)                      // data_size = 2 bytes
                 | (1u << 20)                      // pad_enable
                 | (3u << 22)                      // pad_interval = 16 dwords
                 | (3u << 25)),                    // pad_amount   = 4 dwords
             (int)(tensor_d0 << 16),                       // dim0 lo16
             (int)((tensor_d0 >> 16) | (tensor_d1 << 16)), // dim0 hi / dim1 lo
             (int)((tensor_d1 >> 16) | (tile_d0  << 16)),  // dim1 hi / tile_d0
             (int)tile_d1,                                 // tile_d1, tile_d2=0
             (int)stride0,                                 // dim0 stride lo32
             0, 0 };
  v4i gz = { 0, 0, 0, 0 };
#if __has_include(<hip/amd_detail/amd_gfx1250_TDM.h>)
  v8i gz8 = { 0, 0, 0, 0, 0, 0, 0, 0 };
  __builtin_amdgcn_tensor_load_to_lds(g0, g1, gz, gz, gz8, 0);   // 6-arg toolchain
#else
  __builtin_amdgcn_tensor_load_to_lds(g0, g1, gz, gz, 0);        // 5-arg toolchain
#endif
}
#endif

// ---------------------------------------------------------------------------
// Dense GEMM: Y[N,512](f32) = Abf[N,512](bf16) x Wt[512,512](bf16, [n][k])
// 256 thr = 8 waves; tile 64M x 128N; K-step 32; wave = 16M x 64N.
// TDM path: wave0 drives double-buffered async tile DMA (A 32x64, B 32x128),
// compute waves only do fragment ds_load_b128 + WMMA.
// ---------------------------------------------------------------------------
__global__ __launch_bounds__(256)
void gemm_bf16(const unsigned short* __restrict__ Abf,
               const unsigned short* __restrict__ Wt,
               float* __restrict__ Y) {
  const int tid    = threadIdx.x;
  const int lane   = tid & 31;
  const int wid    = tid >> 5;
  const int waveM  = wid & 3;
  const int waveN  = wid >> 2;
  const int mBlock = blockIdx.y * 64;
  const int nBlock = blockIdx.x * 128;

  v8f acc[4] = {};

  // fragment lane layout: A row m / B col n = lane&15, both with contiguous k = 16*(lane>>4) + s
  const int m    = lane & 15;
  const int kbA  = (lane & 16) ? 16 : 0;
  const int kbB  = (lane & 16) ? 16 : 0;
  const int aoff = (waveM * 16 + m) * LDA + kbA;
  const int boff = (waveN * 64 + m) * LDB + kbB;

#if HAVE_TDM
  __shared__ unsigned short sA[2 * 64 * LDA];    // double-buffered A tiles
  __shared__ unsigned short sB[2 * 128 * LDB];   // double-buffered B tiles

  const unsigned ldsA0 = (unsigned)(size_t)(void*)sA;
  const unsigned ldsB0 = (unsigned)(size_t)(void*)sB;
  const unsigned rowsA = (unsigned)(GN - mBlock);  // TDM OOB rows read as zero

  if (wid == 0) {
    tdm_load_2d(ldsA0, Abf + (size_t)mBlock * GD, GD, rowsA, 32u, 64u,  GD);
    tdm_load_2d(ldsB0, Wt  + (size_t)nBlock * GD, GD, GD,    32u, 128u, GD);
  }

  for (int k0 = 0; k0 < GD; k0 += 32) {
    const int cur = (k0 >> 5) & 1;
    if (wid == 0) {                                 // wave-uniform branch
      if (k0 + 32 < GD) {
        const int nxt = cur ^ 1;
        tdm_load_2d(ldsA0 + (unsigned)(nxt * 64  * LDA * 2),
                    Abf + (size_t)mBlock * GD + (k0 + 32), GD, rowsA, 32u, 64u,  GD);
        tdm_load_2d(ldsB0 + (unsigned)(nxt * 128 * LDB * 2),
                    Wt  + (size_t)nBlock * GD + (k0 + 32), GD, GD,    32u, 128u, GD);
        __builtin_amdgcn_s_wait_tensorcnt(2);       // current pair landed
      } else {
        __builtin_amdgcn_s_wait_tensorcnt(0);       // drain on last tile
      }
    }
    __syncthreads();
    const unsigned short* sAc = sA + cur * (64 * LDA);
    const unsigned short* sBc = sB + cur * (128 * LDB);
    v16bf afrag = mk16(*(const v8us*)(sAc + aoff), *(const v8us*)(sAc + aoff + 8));
    #pragma unroll
    for (int t = 0; t < 4; ++t) {
      const unsigned short* bp = sBc + boff + t * 16 * LDB;
      v16bf bfrag = mk16(*(const v8us*)bp, *(const v8us*)(bp + 8));
      acc[t] = __builtin_amdgcn_wmma_f32_16x16x32_bf16(
          false, afrag, false, bfrag, (short)0, acc[t], false, false);
    }
    __syncthreads();
  }
#else
  // fallback: register-pipelined synchronous staging
  __shared__ unsigned short sA[64 * LDA];
  __shared__ unsigned short sB[128 * LDB];

  const int srow = tid >> 2;
  const int sc   = (tid & 3) * 8;
  int agrow = mBlock + srow; if (agrow > GN - 1) agrow = GN - 1;
  const unsigned short* aSrc  = Abf + (size_t)agrow * GD + sc;
  const unsigned short* bSrc0 = Wt + (size_t)(nBlock + srow) * GD + sc;
  const unsigned short* bSrc1 = Wt + (size_t)(nBlock + srow + 64) * GD + sc;
  uint4* aDst  = (uint4*)&sA[srow * LDA + sc];
  uint4* bDst0 = (uint4*)&sB[srow * LDB + sc];
  uint4* bDst1 = (uint4*)&sB[(srow + 64) * LDB + sc];

  uint4 ra  = *(const uint4*)aSrc;
  uint4 rb0 = *(const uint4*)bSrc0;
  uint4 rb1 = *(const uint4*)bSrc1;

  for (int k0 = 0; k0 < GD; k0 += 32) {
    *aDst = ra; *bDst0 = rb0; *bDst1 = rb1;
    __syncthreads();
    if (k0 + 32 < GD) {
      ra  = *(const uint4*)(aSrc  + k0 + 32);
      rb0 = *(const uint4*)(bSrc0 + k0 + 32);
      rb1 = *(const uint4*)(bSrc1 + k0 + 32);
    }
    v16bf afrag = mk16(*(const v8us*)(sA + aoff), *(const v8us*)(sA + aoff + 8));
    #pragma unroll
    for (int t = 0; t < 4; ++t) {
      const unsigned short* bp = sB + boff + t * 16 * LDB;
      v16bf bfrag = mk16(*(const v8us*)bp, *(const v8us*)(bp + 8));
      acc[t] = __builtin_amdgcn_wmma_f32_16x16x32_bf16(
          false, afrag, false, bfrag, (short)0, acc[t], false, false);
    }
    __syncthreads();
  }
#endif

  // C/D layout: VGPR r -> M = r (+8 upper half-wave), N = lane&15
  const int mw = mBlock + waveM * 16 + ((lane & 16) ? 8 : 0);
  const int nw = nBlock + waveN * 64 + (lane & 15);
  if (mBlock + 64 <= GN) {           // uniform fast path: no per-store guards
    #pragma unroll
    for (int t = 0; t < 4; ++t)
      #pragma unroll
      for (int r = 0; r < 8; ++r)
        Y[(size_t)(mw + r) * GD + nw + t * 16] = acc[t][r];
  } else {                           // tail M-block only (1 of 1563)
    #pragma unroll
    for (int t = 0; t < 4; ++t)
      #pragma unroll
      for (int r = 0; r < 8; ++r)
        if (mw + r < GN) Y[(size_t)(mw + r) * GD + nw + t * 16] = acc[t][r];
  }
}

// ---------------------------------------------------------------------------
// SpMM scatter: acc[row[e],:] += val[e] * Y[col[e],:]  (128 threads / edge)
// ---------------------------------------------------------------------------
__global__ __launch_bounds__(256)
void spmm_scatter(const int* __restrict__ er, const int* __restrict__ ec,
                  const float* __restrict__ ev, const float* __restrict__ Y,
                  float* __restrict__ acc) {
  unsigned int t = blockIdx.x * 256u + threadIdx.x;
  unsigned int e = t >> 7;
  unsigned int c = (t & 127u) << 2;
  if (e >= GE) return;
  int   r  = er[e];
  int   cl = ec[e];
  float v  = ev[e];
  const float4 y = *(const float4*)(Y + (size_t)cl * GD + c);
  float* dst = acc + (size_t)r * GD + c;
  atomicAdd(dst + 0, v * y.x);
  atomicAdd(dst + 1, v * y.y);
  atomicAdd(dst + 2, v * y.z);
  atomicAdd(dst + 3, v * y.w);
}

__global__ __launch_bounds__(256)
void zero_f4(float4* __restrict__ p, int n4) {
  int i = blockIdx.x * 256 + threadIdx.x;
  if (i < n4) p[i] = make_float4(0.f, 0.f, 0.f, 0.f);
}

// f32 -> bf16 bulk convert (for X)
__global__ __launch_bounds__(256)
void cvt_bf16_f4(const float4* __restrict__ in, unsigned short* __restrict__ out,
                 int n4) {
  int i = blockIdx.x * 256 + threadIdx.x;
  if (i >= n4) return;
  *(unsigned long long*)(out + (size_t)i * 4) = pack4bf(in[i]);
}

// convert + transpose weights: Wt[n][k] = bf16(W[k][n]); 512x512, one-off
__global__ __launch_bounds__(256)
void cvt_transpose_w(const float* __restrict__ W, unsigned short* __restrict__ Wt) {
  int idx = blockIdx.x * 256 + threadIdx.x;   // 0..262143
  int k = idx >> 9;
  int n = idx & 511;
  Wt[(size_t)n * GD + k] = f2bf(W[idx]);
}

// tanh, f32 output (feeds l2norm)
__global__ __launch_bounds__(256)
void tanh_f4(const float4* __restrict__ in, float4* __restrict__ out, int n4) {
  int i = blockIdx.x * 256 + threadIdx.x;
  if (i >= n4) return;
  float4 v = in[i];
  out[i] = make_float4(tanhf(v.x), tanhf(v.y), tanhf(v.z), tanhf(v.w));
}

// tanh, bf16 output (h1 feeds only the next GEMM)
__global__ __launch_bounds__(256)
void tanh_f4_bf16(const float4* __restrict__ in, unsigned short* __restrict__ out,
                  int n4) {
  int i = blockIdx.x * 256 + threadIdx.x;
  if (i >= n4) return;
  float4 v = in[i];
  float4 t = make_float4(tanhf(v.x), tanhf(v.y), tanhf(v.z), tanhf(v.w));
  *(unsigned long long*)(out + (size_t)i * 4) = pack4bf(t);
}

// one wave32 per row: sum of squares reduce + scale
__global__ __launch_bounds__(256)
void l2norm_rows(const float* __restrict__ H, float* __restrict__ out) {
  int row  = blockIdx.x * 8 + (threadIdx.x >> 5);
  int lane = threadIdx.x & 31;
  if (row >= GN) return;
  const float4* p = (const float4*)(H + (size_t)row * GD);
  float4 v[4];
  float s = 0.f;
  #pragma unroll
  for (int i = 0; i < 4; ++i) {
    v[i] = p[lane + i * 32];
    s += v[i].x * v[i].x + v[i].y * v[i].y + v[i].z * v[i].z + v[i].w * v[i].w;
  }
  #pragma unroll
  for (int off = 16; off > 0; off >>= 1)
    s += __shfl_xor(s, off, 32);
  float inv = 1.0f / sqrtf(fmaxf(s, 1e-12f));
  float4* o = (float4*)(out + (size_t)row * GD);
  #pragma unroll
  for (int i = 0; i < 4; ++i)
    o[lane + i * 32] = make_float4(v[i].x * inv, v[i].y * inv, v[i].z * inv, v[i].w * inv);
}

// ---------------------------------------------------------------------------
extern "C" void kernel_launch(void* const* d_in, const int* in_sizes, int n_in,
                              void* d_out, int out_size, void* d_ws, size_t ws_size,
                              hipStream_t stream) {
  const float* X  = (const float*)d_in[0];
  const int*   er = (const int*)  d_in[1];
  const int*   ec = (const int*)  d_in[2];
  const float* ev = (const float*)d_in[3];
  const float* W1 = (const float*)d_in[4];
  const float* W2 = (const float*)d_in[5];
  float* out = (float*)d_out;

  // workspace layout (~513 MB):
  float*          ws0 = (float*)d_ws;                             // N*D f32
  float*          ws1 = ws0 + (size_t)GN * GD;                    // N*D f32
  unsigned short* wsb = (unsigned short*)(ws1 + (size_t)GN * GD); // N*D bf16 (X, then h1)
  unsigned short* w1t = wsb + (size_t)GN * GD;                    // 512x512 bf16
  unsigned short* w2t = w1t + (size_t)GD * GD;                    // 512x512 bf16

  const int n4 = GN * GD / 4;                       // 12.8M float4
  dim3 gGemm(GD / 128, (GN + 63) / 64);             // (4, 1563)
  const unsigned int spmmBlocks =
      (unsigned int)(((unsigned long long)GE * 128ull + 255ull) / 256ull);
  const int ewBlocks  = (n4 + 255) / 256;
  const int wBlocks   = (GD * GD + 255) / 256;      // 1024
  const int nrmBlocks = (GN + 7) / 8;

  // one-off conversions
  cvt_bf16_f4    <<<ewBlocks, 256, 0, stream>>>((const float4*)X, wsb, n4);
  cvt_transpose_w<<<wBlocks,  256, 0, stream>>>(W1, w1t);
  cvt_transpose_w<<<wBlocks,  256, 0, stream>>>(W2, w2t);

  // ----- layer 1: h1 = tanh(spmm(X @ W1)) -----
  gemm_bf16   <<<gGemm,      256, 0, stream>>>(wsb, w1t, ws0);
  zero_f4     <<<ewBlocks,   256, 0, stream>>>((float4*)ws1, n4);
  spmm_scatter<<<spmmBlocks, 256, 0, stream>>>(er, ec, ev, ws0, ws1);
  tanh_f4_bf16<<<ewBlocks,   256, 0, stream>>>((const float4*)ws1, wsb, n4); // h1 bf16

  // ----- layer 2: h2 = tanh(spmm(h1 @ W2)) -----
  gemm_bf16   <<<gGemm,      256, 0, stream>>>(wsb, w2t, ws0);
  zero_f4     <<<ewBlocks,   256, 0, stream>>>((float4*)ws1, n4);
  spmm_scatter<<<spmmBlocks, 256, 0, stream>>>(er, ec, ev, ws0, ws1);
  tanh_f4     <<<ewBlocks,   256, 0, stream>>>((const float4*)ws1, (float4*)ws0, n4);

  // ----- out = l2_normalize(h2, axis=1) -----
  l2norm_rows <<<nrmBlocks,  256, 0, stream>>>(ws0, out);
}
